// CrossLinearAttention_49203145343716
// MI455X (gfx1250) — matrix operand order repaired and checked by reading, not compiled, not run on hardware
//
#include <hip/hip_runtime.h>
#include <hip/hip_bf16.h>

// ---------- problem constants ----------
#define BATCH   8
#define N1      4096
#define N2      4096
#define DIMX    256
#define NHEADS  8
#define DH      64
#define INNERD  512
#define TILE    64

#define FS 24   // bf16 slots per lane-fragment (16 used + 8 pad -> 48B stride, bank-conflict free)

typedef __bf16 v16bf __attribute__((ext_vector_type(16)));
typedef __bf16 v8bf  __attribute__((ext_vector_type(8)));
typedef float  v8f   __attribute__((ext_vector_type(8)));

#define WMMA_BF16(A, B, C) \
  __builtin_amdgcn_wmma_f32_16x16x32_bf16(false, (A), false, (B), (short)0, (C), false, false)

// ---------------- fragment-layout staging (ISA 7.12.2) ----------------
// A 16x32 bf16 tile: lane (l + 16*hi) holds row M=l; hi=(k>>3)&1; elem=((k&16)>>1)|(k&7)
__device__ __forceinline__ void stage_a(__bf16* buf, int nchunks, int r, int k, __bf16 val) {
  const int mt = r >> 4, l = r & 15;
  const int chunk = k >> 5, kl = k & 31;
  const int lane = l + 16 * ((kl >> 3) & 1);
  const int elem = ((kl & 16) >> 1) | (kl & 7);
  buf[(size_t)((mt * nchunks + chunk) * 32 + lane) * FS + elem] = val;
}

// B 32x16 bf16 tile: lane (l + 16*hi) holds col N=l; hi=(k>>3)&1; elem=((k&16)>>1)|(k&7)
__device__ __forceinline__ void stage_b(__bf16* buf, int ntiles, int k, int c, __bf16 val) {
  const int nt = c >> 4, l = c & 15;
  const int chunk = k >> 5, kl = k & 31;
  const int lane = l + 16 * ((kl >> 3) & 1);
  const int elem = ((kl & 16) >> 1) | (kl & 7);
  buf[(size_t)((chunk * ntiles + nt) * 32 + lane) * FS + elem] = val;
}

// contiguous per-lane fragment read: two 16B LDS loads
__device__ __forceinline__ v16bf read_frag(const __bf16* p) {
  const v8bf lo = *(const v8bf*)(p);
  const v8bf hi = *(const v8bf*)(p + 8);
  v16bf r;
#pragma unroll
  for (int i = 0; i < 8; ++i) { r[i] = lo[i]; r[i + 8] = hi[i]; }
  return r;
}

__device__ __forceinline__ size_t frag_off(int tileLinear, int lane) {
  return (size_t)(tileLinear * 32 + lane) * FS;
}

// C/D 16x16 f32: VGPR j -> M = j + 8*hi, N = l.
__device__ __forceinline__ void store_c_f32(float* base, int stride,
                                            int row0, int col0, v8f c, int lane) {
  const int l = lane & 15, hi = lane >> 4;
#pragma unroll
  for (int j = 0; j < 8; ++j)
    base[(size_t)(row0 + hi * 8 + j) * stride + col0 + l] = c[j];
}

__device__ __forceinline__ float inv_freq_of(int j) {
  return __powf(10000.0f, -(float)j * (1.0f / 16.0f));  // 1/10000^(j/16)
}

// 2D rotary on a 64-wide head row in LDS (dims 0..31 with p0, 32..63 with p1)
__device__ __forceinline__ void rotary_row(float* row, float p0, float p1) {
#pragma unroll
  for (int j = 0; j < 16; ++j) {
    const float invf = inv_freq_of(j);
    const float f0 = p0 * invf, f1 = p1 * invf;
    const float c0 = __cosf(f0), s0 = __sinf(f0);
    const float c1 = __cosf(f1), s1 = __sinf(f1);
    const float a0 = row[j],      b0 = row[j + 16];
    row[j]      = a0 * c0 - b0 * s0;
    row[j + 16] = b0 * c0 + a0 * s0;
    const float a1 = row[32 + j], b1 = row[48 + j];
    row[32 + j] = a1 * c1 - b1 * s1;
    row[48 + j] = b1 * c1 + a1 * s1;
  }
}

// ================= kernel 0: zero dots workspace =================
__global__ void zero_ws_kernel(float* __restrict__ p, int n) {
  int i = blockIdx.x * blockDim.x + threadIdx.x;
  if (i < n) p[i] = 0.0f;
}

// ================= kernel 1: kv projection + norm + rotary + dots =================
__global__ __launch_bounds__(256) void kv_dots_kernel(const float* __restrict__ z,
                                                      const float* __restrict__ z_pos,
                                                      const float* __restrict__ Wkv,
                                                      float* __restrict__ dots) {
  // fragment-layout staging buffers
  __shared__ __attribute__((aligned(16))) __bf16 zAf[4 * 8 * 32 * FS];  // z tile as A (4 mt x 8 kchunk)
  __shared__ __attribute__((aligned(16))) __bf16 wBf[8 * 4 * 32 * FS];  // Wk/Wv slice as B (8 kchunk x 4 nt)
  __shared__ float kf[TILE][DH + 4];                                    // k_h f32 (norm+rotary)
  __shared__ float vf[TILE][DH + 4];                                    // v_h f32 (norm)
  __shared__ __attribute__((aligned(16))) __bf16 kAf[4 * 2 * 32 * FS];  // k^T as A (4 mt x 2 kchunk)
  __shared__ __attribute__((aligned(16))) __bf16 vBf[2 * 4 * 32 * FS];  // v as B (2 kchunk x 4 nt)

  const int tiles = N2 / TILE;
  const int b  = blockIdx.x / tiles;
  const int n0 = (blockIdx.x % tiles) * TILE;
  const int tid  = threadIdx.x;
  const int wave = tid >> 5, lane = tid & 31;
  const int mt  = wave & 3;           // M tile
  const int nt0 = (wave >> 2) * 2;    // first of 2 N tiles

  // stage z tile (64 x 256) directly into A-fragment layout
  for (int i = tid; i < TILE * DIMX; i += 256) {
    const int r = i >> 8, c = i & 255;
    stage_a(zAf, 8, r, c, (__bf16)z[(size_t)(b * N2 + n0 + r) * DIMX + c]);
  }
  __syncthreads();

  for (int h = 0; h < NHEADS; ++h) {
    // ---- stage Wk_h (B layout) ; k_h = z_tile @ Wk_h ----
    for (int i = tid; i < DIMX * DH; i += 256) {
      const int r = i >> 6, c = i & 63;
      stage_b(wBf, 4, r, c, (__bf16)Wkv[(size_t)r * (2 * INNERD) + h * DH + c]);
    }
    if (h + 1 < NHEADS)
      __builtin_prefetch(&Wkv[(size_t)tid * (2 * INNERD) + (h + 1) * DH], 0, 0);
    __syncthreads();
    {
      v8f a0 = {0.f,0.f,0.f,0.f,0.f,0.f,0.f,0.f}, a1 = a0;
#pragma unroll
      for (int kk = 0; kk < 8; ++kk) {
        const v16bf a = read_frag(&zAf[frag_off(mt * 8 + kk, lane)]);
        a0 = WMMA_BF16(a, read_frag(&wBf[frag_off(kk * 4 + nt0,     lane)]), a0);
        a1 = WMMA_BF16(a, read_frag(&wBf[frag_off(kk * 4 + nt0 + 1, lane)]), a1);
      }
      store_c_f32(&kf[0][0], DH + 4, mt * 16, nt0 * 16,       a0, lane);
      store_c_f32(&kf[0][0], DH + 4, mt * 16, (nt0 + 1) * 16, a1, lane);
    }
    __syncthreads();

    // ---- stage Wv_h ; v_h = z_tile @ Wv_h ----
    for (int i = tid; i < DIMX * DH; i += 256) {
      const int r = i >> 6, c = i & 63;
      stage_b(wBf, 4, r, c, (__bf16)Wkv[(size_t)r * (2 * INNERD) + INNERD + h * DH + c]);
    }
    __syncthreads();
    {
      v8f a0 = {0.f,0.f,0.f,0.f,0.f,0.f,0.f,0.f}, a1 = a0;
#pragma unroll
      for (int kk = 0; kk < 8; ++kk) {
        const v16bf a = read_frag(&zAf[frag_off(mt * 8 + kk, lane)]);
        a0 = WMMA_BF16(a, read_frag(&wBf[frag_off(kk * 4 + nt0,     lane)]), a0);
        a1 = WMMA_BF16(a, read_frag(&wBf[frag_off(kk * 4 + nt0 + 1, lane)]), a1);
      }
      store_c_f32(&vf[0][0], DH + 4, mt * 16, nt0 * 16,       a0, lane);
      store_c_f32(&vf[0][0], DH + 4, mt * 16, (nt0 + 1) * 16, a1, lane);
    }
    __syncthreads();

    // ---- instance norm over last dim (64) for k and v ----
    if (tid < 128) {
      float* row = (tid < 64) ? &kf[tid][0] : &vf[tid - 64][0];
      float mu = 0.f;
#pragma unroll
      for (int j = 0; j < DH; ++j) mu += row[j];
      mu *= (1.0f / DH);
      float var = 0.f;
#pragma unroll
      for (int j = 0; j < DH; ++j) { const float d = row[j] - mu; var += d * d; }
      var *= (1.0f / DH);
      const float rs = rsqrtf(var + 1e-5f);
#pragma unroll
      for (int j = 0; j < DH; ++j) row[j] = (row[j] - mu) * rs;
    }
    __syncthreads();

    // ---- 2D rotary on k rows ----
    if (tid < TILE) {
      const float p0 = z_pos[(size_t)(b * N2 + n0 + tid) * 2 + 0] * 64.0f;  // SCALE/MIN_FREQ
      const float p1 = z_pos[(size_t)(b * N2 + n0 + tid) * 2 + 1] * 64.0f;
      rotary_row(&kf[tid][0], p0, p1);
    }
    __syncthreads();

    // ---- restage: k^T as A fragments, v as B fragments ----
    for (int i = tid; i < TILE * DH; i += 256) {
      const int n = i >> 6, d = i & 63;
      stage_a(kAf, 2, d, n, (__bf16)kf[n][d]);   // A[d][n] = k[n][d]
      stage_b(vBf, 4, n, d, (__bf16)vf[n][d]);   // B[n][e]
    }
    __syncthreads();

    // ---- dots_h += k_h^T @ v_h ----
    {
      v8f a0 = {0.f,0.f,0.f,0.f,0.f,0.f,0.f,0.f}, a1 = a0;
#pragma unroll
      for (int kk = 0; kk < 2; ++kk) {
        const v16bf a = read_frag(&kAf[frag_off(mt * 2 + kk, lane)]);
        a0 = WMMA_BF16(a, read_frag(&vBf[frag_off(kk * 4 + nt0,     lane)]), a0);
        a1 = WMMA_BF16(a, read_frag(&vBf[frag_off(kk * 4 + nt0 + 1, lane)]), a1);
      }
      float* D = dots + (size_t)(b * NHEADS + h) * DH * DH;
      const int l = lane & 15, hi = lane >> 4;
#pragma unroll
      for (int j = 0; j < 8; ++j) {
        unsafeAtomicAdd(&D[(size_t)(mt * 16 + hi * 8 + j) * DH + nt0 * 16 + l],       a0[j]);
        unsafeAtomicAdd(&D[(size_t)(mt * 16 + hi * 8 + j) * DH + (nt0 + 1) * 16 + l], a1[j]);
      }
    }
    __syncthreads();
  }
}

// ================= kernel 2: q projection + rotary + q@dots + @Wout + bias =================
__global__ __launch_bounds__(256) void q_out_kernel(const float* __restrict__ x,
                                                    const float* __restrict__ x_pos,
                                                    const float* __restrict__ Wq,
                                                    const float* __restrict__ Wout,
                                                    const float* __restrict__ bout,
                                                    const float* __restrict__ dots,
                                                    float* __restrict__ out) {
  __shared__ __attribute__((aligned(16))) __bf16 xAf[4 * 8 * 32 * FS];    // x tile as A
  __shared__ __attribute__((aligned(16))) __bf16 wFrag[32 * 32 * FS];     // Wq_h B (8x4) / Wout_h B (2x16)
  __shared__ float qf[TILE][DH + 4];                                      // q_h f32 (rotary)
  __shared__ __attribute__((aligned(16))) __bf16 qAf[4 * 2 * 32 * FS];    // q as A
  __shared__ __attribute__((aligned(16))) __bf16 dBf[2 * 4 * 32 * FS];    // dots_h as B
  __shared__ __attribute__((aligned(16))) __bf16 tAf[4 * 2 * 32 * FS];    // (q@dots)/n2 as A

  const int tiles = N1 / TILE;
  const int b  = blockIdx.x / tiles;
  const int n0 = (blockIdx.x % tiles) * TILE;
  const int tid  = threadIdx.x;
  const int wave = tid >> 5, lane = tid & 31;
  const int mt   = wave & 3;
  const int nt0  = (wave >> 2) * 2;
  const int ntF0 = (wave >> 2) * 8;   // final 64x256: 8 N tiles per wave
  const int l = lane & 15, hi = lane >> 4;

  v8f accF[8];
#pragma unroll
  for (int i = 0; i < 8; ++i) accF[i] = (v8f){0.f,0.f,0.f,0.f,0.f,0.f,0.f,0.f};

  for (int i = tid; i < TILE * DIMX; i += 256) {
    const int r = i >> 8, c = i & 255;
    stage_a(xAf, 8, r, c, (__bf16)x[(size_t)(b * N1 + n0 + r) * DIMX + c]);
  }
  __syncthreads();

  for (int h = 0; h < NHEADS; ++h) {
    // ---- stage Wq_h (B 8x4) ; q_h = x_tile @ Wq_h ----
    for (int i = tid; i < DIMX * DH; i += 256) {
      const int r = i >> 6, c = i & 63;
      stage_b(wFrag, 4, r, c, (__bf16)Wq[(size_t)r * INNERD + h * DH + c]);
    }
    __syncthreads();
    {
      v8f a0 = {0.f,0.f,0.f,0.f,0.f,0.f,0.f,0.f}, a1 = a0;
#pragma unroll
      for (int kk = 0; kk < 8; ++kk) {
        const v16bf a = read_frag(&xAf[frag_off(mt * 8 + kk, lane)]);
        a0 = WMMA_BF16(a, read_frag(&wFrag[frag_off(kk * 4 + nt0,     lane)]), a0);
        a1 = WMMA_BF16(a, read_frag(&wFrag[frag_off(kk * 4 + nt0 + 1, lane)]), a1);
      }
      store_c_f32(&qf[0][0], DH + 4, mt * 16, nt0 * 16,       a0, lane);
      store_c_f32(&qf[0][0], DH + 4, mt * 16, (nt0 + 1) * 16, a1, lane);
    }
    __syncthreads();

    // ---- rotary on q rows (no norm for q) ----
    if (tid < TILE) {
      const float p0 = x_pos[(size_t)(b * N1 + n0 + tid) * 2 + 0] * 64.0f;
      const float p1 = x_pos[(size_t)(b * N1 + n0 + tid) * 2 + 1] * 64.0f;
      rotary_row(&qf[tid][0], p0, p1);
    }
    // ---- stage dots_h as B fragments (2x4) ----
    for (int i = tid; i < DH * DH; i += 256) {
      const int r = i >> 6, c = i & 63;
      stage_b(dBf, 4, r, c, (__bf16)dots[(size_t)(b * NHEADS + h) * DH * DH + i]);
    }
    __syncthreads();

    // ---- restage rotated q as A fragments ----
    for (int i = tid; i < TILE * DH; i += 256) {
      const int r = i >> 6, c = i & 63;
      stage_a(qAf, 2, r, c, (__bf16)qf[r][c]);
    }
    __syncthreads();

    // ---- t_h = (q_h @ dots_h)/n2 ; write straight into A-fragment layout ----
    {
      v8f a0 = {0.f,0.f,0.f,0.f,0.f,0.f,0.f,0.f}, a1 = a0;
#pragma unroll
      for (int kk = 0; kk < 2; ++kk) {
        const v16bf a = read_frag(&qAf[frag_off(mt * 2 + kk, lane)]);
        a0 = WMMA_BF16(a, read_frag(&dBf[frag_off(kk * 4 + nt0,     lane)]), a0);
        a1 = WMMA_BF16(a, read_frag(&dBf[frag_off(kk * 4 + nt0 + 1, lane)]), a1);
      }
      const float s = 1.0f / (float)N2;
#pragma unroll
      for (int j = 0; j < 8; ++j) {
        stage_a(tAf, 2, mt * 16 + hi * 8 + j, nt0 * 16 + l,       (__bf16)(a0[j] * s));
        stage_a(tAf, 2, mt * 16 + hi * 8 + j, (nt0 + 1) * 16 + l, (__bf16)(a1[j] * s));
      }
    }
    __syncthreads();

    // ---- stage Wout rows [h*64, h*64+64) as B fragments (2x16, reuses wFrag) ----
    for (int i = tid; i < DH * DIMX; i += 256) {
      const int r = i >> 8, c = i & 255;
      stage_b(wFrag, 16, r, c, (__bf16)Wout[(size_t)(h * DH + r) * DIMX + c]);
    }
    __syncthreads();

    // ---- out_tile += t_h @ Wout_h (A reused across 8 N tiles) ----
#pragma unroll
    for (int kk = 0; kk < 2; ++kk) {
      const v16bf a = read_frag(&tAf[frag_off(mt * 2 + kk, lane)]);
#pragma unroll
      for (int i = 0; i < 8; ++i)
        accF[i] = WMMA_BF16(a, read_frag(&wFrag[frag_off(kk * 16 + ntF0 + i, lane)]), accF[i]);
    }
    __syncthreads();
  }

  // ---- epilogue: add bias, store ----
#pragma unroll
  for (int i = 0; i < 8; ++i) {
    const int col0 = (ntF0 + i) * 16;
    const float bias = bout[col0 + l];
#pragma unroll
    for (int j = 0; j < 8; ++j) {
      const int m = mt * 16 + hi * 8 + j;
      out[(size_t)(b * N1 + n0 + m) * DIMX + col0 + l] = accF[i][j] + bias;
    }
  }
}

// ================= host launcher =================
extern "C" void kernel_launch(void* const* d_in, const int* in_sizes, int n_in,
                              void* d_out, int out_size, void* d_ws, size_t ws_size,
                              hipStream_t stream) {
  const float* x     = (const float*)d_in[0];
  const float* z     = (const float*)d_in[1];
  const float* x_pos = (const float*)d_in[2];
  const float* z_pos = (const float*)d_in[3];
  const float* Wq    = (const float*)d_in[4];
  const float* Wkv   = (const float*)d_in[5];
  const float* Wout  = (const float*)d_in[6];
  const float* bout  = (const float*)d_in[7];
  float* out  = (float*)d_out;
  float* dots = (float*)d_ws;   // BATCH*NHEADS*DH*DH floats = 1 MB

  const int ndots = BATCH * NHEADS * DH * DH;
  zero_ws_kernel<<<(ndots + 255) / 256, 256, 0, stream>>>(dots, ndots);
  kv_dots_kernel<<<BATCH * (N2 / TILE), 256, 0, stream>>>(z, z_pos, Wkv, dots);
  q_out_kernel<<<BATCH * (N1 / TILE), 256, 0, stream>>>(x, x_pos, Wq, Wout, bout, dots, out);
}
